// Attention_28750511080172
// MI455X (gfx1250) — compile-verified, matrix operand order checked by reading, not run on hardware
//
#include <hip/hip_runtime.h>
#include <hip/hip_bf16.h>

typedef __attribute__((ext_vector_type(16))) _Float16 v16h;
typedef __attribute__((ext_vector_type(8)))  _Float16 v8h;
typedef __attribute__((ext_vector_type(8)))  float    v8f;

#define DEV __device__ __forceinline__

constexpr int Bsz = 2, S = 2048, D = 1024, H = 16, DEPTH = 64, M = Bsz * S;

DEV v16h mk_frag(v8h lo, v8h hi) {
  return __builtin_shufflevector(lo, hi, 0,1,2,3,4,5,6,7,8,9,10,11,12,13,14,15);
}

DEV v8f wmma_f16(v16h a, v16h b, v8f c) {
  // D = A(16x32 f16) * B(32x16 f16) + C(16x16 f32)
  return __builtin_amdgcn_wmma_f32_16x16x32_f16(false, a, false, b, (short)0, c,
                                                false, false);
}

// ---- DPP16 butterfly reductions over each 16-lane half (VALU only) --------
// xor1 = quad_perm(1,0,3,2)=0xB1, xor2 = quad_perm(2,3,0,1)=0x4E,
// then row_half_mirror (0x141) folds quads->octs, row_mirror (0x140) octs->row.
DEV float dpp_bfly(float x, int ctrl_is_unused) { return x; }  // (unused)

DEV float row_max16(float x) {
  int t;
  t = __builtin_amdgcn_update_dpp(0, __float_as_int(x), 0xB1, 0xF, 0xF, true);
  x = fmaxf(x, __int_as_float(t));
  t = __builtin_amdgcn_update_dpp(0, __float_as_int(x), 0x4E, 0xF, 0xF, true);
  x = fmaxf(x, __int_as_float(t));
  t = __builtin_amdgcn_update_dpp(0, __float_as_int(x), 0x141, 0xF, 0xF, true);
  x = fmaxf(x, __int_as_float(t));
  t = __builtin_amdgcn_update_dpp(0, __float_as_int(x), 0x140, 0xF, 0xF, true);
  x = fmaxf(x, __int_as_float(t));
  return x;
}

DEV float row_sum16(float x) {
  int t;
  t = __builtin_amdgcn_update_dpp(0, __float_as_int(x), 0xB1, 0xF, 0xF, true);
  x += __int_as_float(t);
  t = __builtin_amdgcn_update_dpp(0, __float_as_int(x), 0x4E, 0xF, 0xF, true);
  x += __int_as_float(t);
  t = __builtin_amdgcn_update_dpp(0, __float_as_int(x), 0x141, 0xF, 0xF, true);
  x += __int_as_float(t);
  t = __builtin_amdgcn_update_dpp(0, __float_as_int(x), 0x140, 0xF, 0xF, true);
  x += __int_as_float(t);
  return x;
}

// ---------------------------------------------------------------------------
// Tiled WMMA GEMM:  C[M x 1024] = A[M x 1024] * W[1024 x 1024]   (K-step 32)
// Workgroup tile 128x64, 8 waves, each wave owns 16 rows x 64 cols (4 accums).
// ---------------------------------------------------------------------------
template <bool A_F16, bool OUT_SPLIT>
__global__ __launch_bounds__(256) void gemm_kernel(const void* __restrict__ Ap,
                                                   const float* __restrict__ W,
                                                   void* __restrict__ Op,
                                                   float scale) {
  __shared__ _Float16 As[128][40];  // 128 rows x 32 k (padded)
  __shared__ _Float16 Bt[64][40];   // W tile transposed: [n][k] (padded)

  const int tid  = threadIdx.x;
  const int lane = tid & 31;
  const int wave = tid >> 5;
  const int N0   = blockIdx.x * 64;
  const int M0   = blockIdx.y * 128;
  const int cn   = lane & 15;         // fragment row/col within 16
  const int kb   = (lane >> 4) * 8;   // K-half base per lane half
  const int rb   = (lane >> 4) * 8;   // C-layout row base per lane half

  v8f acc[4] = {{}, {}, {}, {}};

  for (int k0 = 0; k0 < D; k0 += 32) {
    // ---- stage A tile 128x32 (convert fp32 -> f16 if needed) ----
    {
      const int r = tid >> 1, c = (tid & 1) * 16;
      if constexpr (A_F16) {
        const _Float16* A = (const _Float16*)Ap + (size_t)(M0 + r) * D + k0 + c;
        *(v8h*)&As[r][c]     = *(const v8h*)(A);
        *(v8h*)&As[r][c + 8] = *(const v8h*)(A + 8);
      } else {
        const float* A = (const float*)Ap + (size_t)(M0 + r) * D + k0 + c;
#pragma unroll
        for (int j = 0; j < 4; ++j) {
          float4 f = ((const float4*)A)[j];
          _Float16* d = &As[r][c + j * 4];
          d[0] = (_Float16)f.x; d[1] = (_Float16)f.y;
          d[2] = (_Float16)f.z; d[3] = (_Float16)f.w;
        }
      }
    }
    // ---- stage W tile 32x64, transposed into LDS as [n][k] ----
    {
      const int kr = tid >> 3, cg = (tid & 7) * 8;
      const float* Wp = W + (size_t)(k0 + kr) * D + N0 + cg;
#pragma unroll
      for (int j = 0; j < 8; ++j) Bt[cg + j][kr] = (_Float16)Wp[j];
    }
    __syncthreads();

    // ---- 4 WMMAs per wave per K-step ----
    v16h a = mk_frag(*(const v8h*)&As[wave * 16 + cn][kb],
                     *(const v8h*)&As[wave * 16 + cn][kb + 16]);
#pragma unroll
    for (int nt = 0; nt < 4; ++nt) {
      v16h b = mk_frag(*(const v8h*)&Bt[nt * 16 + cn][kb],
                       *(const v8h*)&Bt[nt * 16 + cn][kb + 16]);
      acc[nt] = wmma_f16(a, b, acc[nt]);
    }
    __syncthreads();
  }

  // ---- epilogue ----
#pragma unroll
  for (int nt = 0; nt < 4; ++nt) {
#pragma unroll
    for (int g = 0; g < 8; ++g) {
      float v = acc[nt][g] * scale;
      int m = M0 + wave * 16 + rb + g;
      int n = N0 + nt * 16 + cn;
      if constexpr (OUT_SPLIT) {
        int bb = m >> 11, ss = m & (S - 1);
        int hh = n >> 6,  dd = n & (DEPTH - 1);
        ((_Float16*)Op)[(((size_t)bb * H + hh) * S + ss) * DEPTH + dd] =
            (_Float16)v;
      } else {
        ((float*)Op)[(size_t)m * D + n] = v;
      }
    }
  }
}

// ---------------------------------------------------------------------------
// Flash attention: per block 128 queries (8 waves x 16), one (b,h) pair.
// kv loop step 32; K tile staged via CDNA5 async global->LDS DMA (ASYNCcnt),
// V tile transposed through VGPRs. All matmuls on v_wmma_f32_16x16x32_f16.
// ---------------------------------------------------------------------------
__global__ __launch_bounds__(256) void flash_kernel(
    const _Float16* __restrict__ Q, const _Float16* __restrict__ K,
    const _Float16* __restrict__ V, const float* __restrict__ bias,
    _Float16* __restrict__ Oa) {
  __shared__ _Float16 Ks[32][72];      // K tile [kv][d]
  __shared__ _Float16 Vt[64][40];      // V tile transposed [d][kv]
  __shared__ _Float16 Ps[8][16][40];   // per-wave P staging (C->A transpose)

  const int tid  = threadIdx.x;
  const int lane = tid & 31;
  const int wave = tid >> 5;
  const int bh   = blockIdx.y;              // b*H + h
  const int q0   = blockIdx.x * 128;
  const int cn   = lane & 15;
  const int kb   = (lane >> 4) * 8;
  const int rb   = (lane >> 4) * 8;
  const int qr   = q0 + wave * 16;          // wave's first query row

  // Q A-fragments (held in registers across the whole kv loop)
  const _Float16* Qrow = Q + ((size_t)bh * S + qr + cn) * DEPTH;
  v16h qf0 = mk_frag(*(const v8h*)(Qrow + kb), *(const v8h*)(Qrow + kb + 16));
  v16h qf1 = mk_frag(*(const v8h*)(Qrow + 32 + kb),
                     *(const v8h*)(Qrow + 32 + kb + 16));

  float m_i[8], l_i[8];
  v8f o[4] = {{}, {}, {}, {}};
#pragma unroll
  for (int g = 0; g < 8; ++g) { m_i[g] = -3.0e38f; l_i[g] = 0.f; }

  const int skr = tid >> 3, sdg = (tid & 7) * 8;  // staging coords
  const _Float16* Kbase = K + (size_t)bh * S * DEPTH;
  const _Float16* Vbase = V + (size_t)bh * S * DEPTH;
  // LDS byte offset of this thread's K-tile destination (generic VA low bits)
  const unsigned kdst = (unsigned)(uintptr_t)&Ks[skr][sdg];

  for (int kv0 = 0; kv0 < S; kv0 += 32) {
    // ---- stage K tile via async global->LDS DMA (16B per thread) ----
    {
      const _Float16* gK = Kbase + (size_t)(kv0 + skr) * DEPTH + sdg;
      asm volatile("global_load_async_to_lds_b128 %0, %1, off"
                   :: "v"(kdst), "v"(gK)
                   : "memory");
    }
    // ---- stage V tile transposed through VGPRs ----
    v8h vv = *(const v8h*)(Vbase + (size_t)(kv0 + skr) * DEPTH + sdg);
#pragma unroll
    for (int j = 0; j < 8; ++j) Vt[sdg + j][skr] = vv[j];
    asm volatile("s_wait_asynccnt 0" ::: "memory");
    __syncthreads();

    // ---- scores: S[16q x 32kv] = Q * K^T (contraction over depth 64) ----
    v8f s0 = {}, s1 = {};
    {
      v16h b00 = mk_frag(*(const v8h*)&Ks[cn][kb],
                         *(const v8h*)&Ks[cn][kb + 16]);
      v16h b01 = mk_frag(*(const v8h*)&Ks[cn][32 + kb],
                         *(const v8h*)&Ks[cn][32 + kb + 16]);
      v16h b10 = mk_frag(*(const v8h*)&Ks[16 + cn][kb],
                         *(const v8h*)&Ks[16 + cn][kb + 16]);
      v16h b11 = mk_frag(*(const v8h*)&Ks[16 + cn][32 + kb],
                         *(const v8h*)&Ks[16 + cn][32 + kb + 16]);
      s0 = wmma_f16(qf0, b00, s0);
      s0 = wmma_f16(qf1, b01, s0);
      s1 = wmma_f16(qf0, b10, s1);
      s1 = wmma_f16(qf1, b11, s1);
    }

    // ---- bias + online softmax (DPP row reductions, VALU only) ----
#pragma unroll
    for (int g = 0; g < 8; ++g) {
      float x0 = s0[g] + bias[(size_t)(qr + rb + g) * S + kv0 + cn];
      float x1 = s1[g] + bias[(size_t)(qr + rb + g) * S + kv0 + 16 + cn];
      float mx = row_max16(fmaxf(x0, x1));
      float mnew = fmaxf(m_i[g], mx);
      float corr = __expf(m_i[g] - mnew);
      float p0 = __expf(x0 - mnew);
      float p1 = __expf(x1 - mnew);
      float rs = row_sum16(p0 + p1);
      l_i[g] = l_i[g] * corr + rs;
      m_i[g] = mnew;
      o[0][g] *= corr; o[1][g] *= corr; o[2][g] *= corr; o[3][g] *= corr;
      Ps[wave][rb + g][cn]      = (_Float16)p0;   // C-layout -> LDS
      Ps[wave][rb + g][16 + cn] = (_Float16)p1;
    }
    asm volatile("s_wait_dscnt 0" ::: "memory");  // wave-local LDS RAW

    // ---- O += P * V (contraction over 32 kv) ----
    v16h pf = mk_frag(*(const v8h*)&Ps[wave][cn][kb],
                      *(const v8h*)&Ps[wave][cn][kb + 16]);
#pragma unroll
    for (int nt = 0; nt < 4; ++nt) {
      v16h bv = mk_frag(*(const v8h*)&Vt[nt * 16 + cn][kb],
                        *(const v8h*)&Vt[nt * 16 + cn][kb + 16]);
      o[nt] = wmma_f16(pf, bv, o[nt]);
    }
    __syncthreads();
  }

  // ---- normalize + store merged-head f16 [B*S][D] ----
  const int bb = bh >> 4, hh = bh & 15;
#pragma unroll
  for (int g = 0; g < 8; ++g) {
    float inv = 1.0f / l_i[g];
    int qq = qr + rb + g;
    size_t base = ((size_t)bb * S + qq) * D + hh * DEPTH;
#pragma unroll
    for (int nt = 0; nt < 4; ++nt)
      Oa[base + nt * 16 + cn] = (_Float16)(o[nt][g] * inv);
  }
}

extern "C" void kernel_launch(void* const* d_in, const int* in_sizes, int n_in,
                              void* d_out, int out_size, void* d_ws,
                              size_t ws_size, hipStream_t stream) {
  (void)in_sizes; (void)n_in; (void)out_size; (void)ws_size;
  const float* x    = (const float*)d_in[0];
  const float* y    = (const float*)d_in[1];
  const float* bias = (const float*)d_in[2];
  const float* Wq   = (const float*)d_in[3];
  const float* Wk   = (const float*)d_in[4];
  const float* Wv   = (const float*)d_in[5];
  const float* Wo   = (const float*)d_in[6];

  _Float16* Qh = (_Float16*)d_ws;            // [B,H,S,64] f16  (8 MB)
  _Float16* Kh = Qh + (size_t)M * D;         // [B,H,S,64] f16  (8 MB)
  _Float16* Vh = Kh + (size_t)M * D;         // [B,H,S,64] f16  (8 MB)
  _Float16* Ah = Vh + (size_t)M * D;         // [B,S,D] merged  (8 MB)

  dim3 blk(256);
  dim3 gproj(D / 64, M / 128);  // (16, 32)
  // q = (x@Wq) * sqrt(depth) folded into epilogue scale
  gemm_kernel<false, true><<<gproj, blk, 0, stream>>>(x, Wq, Qh, 8.0f);
  gemm_kernel<false, true><<<gproj, blk, 0, stream>>>(y, Wk, Kh, 1.0f);
  gemm_kernel<false, true><<<gproj, blk, 0, stream>>>(y, Wv, Vh, 1.0f);

  dim3 gfa(S / 128, Bsz * H);   // (16, 32)
  flash_kernel<<<gfa, blk, 0, stream>>>(Qh, Kh, Vh, bias, Ah);

  gemm_kernel<true, false><<<gproj, blk, 0, stream>>>(Ah, Wo, d_out, 1.0f);
}
